// deterministic_layer_77464030151326
// MI455X (gfx1250) — matrix operand order repaired and checked by reading, not compiled, not run on hardware
//
#include <hip/hip_runtime.h>
#include <stdint.h>

#define NUM_LAYERS 8
#define DIM 5
#define COND 5
#define H 128
#define SPLIT1 2
#define SPLIT2 3
#define N_HID 3
#define BATCH 32768
#define TILE 128
#define THREADS 256

// staged-weight workspace layout (ushort units, per subnet)
#define SUB_W0_OFF 0            // 128 x 32 padded W0
#define SUB_WH_OFF 4096         // 3 x (128 x 128)
#define SUB_WO_OFF 53248        // 16 x 128 padded Wo
#define SUB_TOTAL  55296        // ushorts per subnet (108 KB)
#define N_SUBNETS  (NUM_LAYERS * 4)

typedef __attribute__((ext_vector_type(16))) __bf16 v16bf;
typedef __attribute__((ext_vector_type(8)))  float  v8f;

// ---------- helpers ----------

// f32 -> bf16 bits via native conversion (v_cvt_*_bf16 on gfx1250)
__device__ __forceinline__ unsigned short f2bf(float x) {
  __bf16 h = (__bf16)x;
  return __builtin_bit_cast(unsigned short, h);
}

__device__ __forceinline__ float fast_tanh(float x) {
#if __has_builtin(__builtin_amdgcn_tanhf)
  return __builtin_amdgcn_tanhf(x);   // v_tanh_f32 (CDNA5 TRANS op)
#else
  return tanhf(x);
#endif
}

__device__ __forceinline__ void wait_async0() {
#if __has_builtin(__builtin_amdgcn_s_wait_asynccnt)
  __builtin_amdgcn_s_wait_asynccnt(0);
#else
  asm volatile("s_wait_asynccnt 0" ::: "memory");
#endif
}

// Async DMA: global bf16 (pre-staged layout) -> LDS, 16B per lane per issue.
// GVS mode: saddr = 64-bit base SGPR pair, vaddr = 32-bit offset, vdst = LDS addr.
__device__ __forceinline__ void async_copy(const unsigned short* __restrict__ gsrc,
                                           unsigned short* ldst, int nbytes) {
  unsigned lbase = (unsigned)(size_t)(void*)ldst;  // low 32 bits = LDS offset
  for (int ofs = threadIdx.x * 16; ofs < nbytes; ofs += THREADS * 16) {
    unsigned laddr = lbase + (unsigned)ofs;
    unsigned goff  = (unsigned)ofs;
    asm volatile("global_load_async_to_lds_b128 %0, %1, %2"
                 :: "v"(laddr), "v"(goff), "s"(gsrc)
                 : "memory");
  }
}

union FragU { v16bf v; uint4 q[2]; };

// A fragment: 16x32 bf16 tile from row-major [M][K] LDS buffer.
__device__ __forceinline__ v16bf load_frag_A(const unsigned short* base, int stride,
                                             int m0, int k0, int lane) {
  const int half = lane >> 4;
  const int mr   = lane & 15;
  const unsigned short* p = base + (m0 + mr) * stride + k0 + half * 8;
  FragU f;
  f.q[0] = *(const uint4*)(p);
  f.q[1] = *(const uint4*)(p + 16);
  return f.v;
}

// B fragment: 32x16 bf16 tile, B[k][n] = W[n][k], W row-major [N][K] in LDS.
__device__ __forceinline__ v16bf load_frag_B(const unsigned short* wbase, int stride,
                                             int n0, int k0, int lane) {
  const int half = lane >> 4;
  const int nr   = lane & 15;
  const unsigned short* p = wbase + (n0 + nr) * stride + k0 + half * 8;
  FragU f;
  f.q[0] = *(const uint4*)(p);
  f.q[1] = *(const uint4*)(p + 16);
  return f.v;
}

__device__ __forceinline__ v8f wmma_bf16(v16bf a, v16bf b, v8f c) {
  return __builtin_amdgcn_wmma_f32_16x16x32_bf16(false, a, false, b, (short)0, c,
                                                 false, false);
}

__device__ __forceinline__ void stage_bias(const float* __restrict__ g, int n,
                                           float* lds) {
  for (int i = threadIdx.x; i < H; i += THREADS) lds[i] = (i < n) ? g[i] : 0.f;
}

// ---------- GEMM stages (forceinline: all LDS ptrs must trace to __shared__) ----

// out[r][n] = tanh(sum_k A[r][k]*W[n][k] + bias[n]); M=128,N=128,K=32*KSTEPS
template <int KSTEPS>
__device__ __forceinline__ void dense_tanh(const unsigned short* A, int astride,
                                           const unsigned short* W,
                                           const float* bias,
                                           unsigned short* out) {
  const int lane = threadIdx.x & 31;
  const int wave = threadIdx.x >> 5;
  const int m0   = wave * 16;
  const int half = lane >> 4;
  const int nr   = lane & 15;
  const v8f zero = {0.f, 0.f, 0.f, 0.f, 0.f, 0.f, 0.f, 0.f};
  v8f acc[8];
#pragma unroll
  for (int nt = 0; nt < 8; ++nt) acc[nt] = zero;
#pragma unroll
  for (int ks = 0; ks < KSTEPS; ++ks) {
    const int k0 = ks * 32;
    v16bf a = load_frag_A(A, astride, m0, k0, lane);
#pragma unroll
    for (int nt = 0; nt < 8; ++nt) {
      v16bf b = load_frag_B(W, H, nt * 16, k0, lane);
      acc[nt] = wmma_bf16(a, b, acc[nt]);
    }
  }
#pragma unroll
  for (int nt = 0; nt < 8; ++nt) {
    const int n  = nt * 16 + nr;
    const float bv = bias[n];
#pragma unroll
    for (int j = 0; j < 8; ++j) {
      const int row = m0 + j + half * 8;
      out[row * H + n] = f2bf(fast_tanh(acc[nt][j] + bv));
    }
  }
}

// out[r][n] = sum_k A[r][k]*Wo[n][k] + bias[n], n < out_dim (<=3), f32 result
__device__ __forceinline__ void dense_out(const unsigned short* A,
                                          const unsigned short* W16,
                                          const float* bias, float* out,
                                          int out_dim) {
  const int lane = threadIdx.x & 31;
  const int wave = threadIdx.x >> 5;
  const int m0   = wave * 16;
  const int half = lane >> 4;
  const int n    = lane & 15;
  v8f acc = {0.f, 0.f, 0.f, 0.f, 0.f, 0.f, 0.f, 0.f};
#pragma unroll
  for (int ks = 0; ks < 4; ++ks) {
    v16bf a = load_frag_A(A, H, m0, ks * 32, lane);
    v16bf b = load_frag_B(W16, H, 0, ks * 32, lane);
    acc = wmma_bf16(a, b, acc);
  }
  if (n < out_dim) {
    const float bv = bias[n];
#pragma unroll
    for (int j = 0; j < 8; ++j) {
      const int row = m0 + j + half * 8;
      out[row * 3 + n] = acc[j] + bv;
    }
  }
}

// ---------- one subnet, software-pipelined weight prefetch ----------
// phases: [pf W0] | [pf Wh0, gemm0] | [pf Wh1, h0] | [pf Wh2, h1] | [pf Wo, h2] | [out]

__device__ __forceinline__ void subnet(
    const unsigned short* __restrict__ wsub, const float* __restrict__ b0,
    const float* __restrict__ bh, const float* __restrict__ bo, int out_dim,
    const unsigned short* ubuf, unsigned short* actA, unsigned short* actB,
    unsigned short* wbA, unsigned short* wbB, float* biasA, float* biasB,
    float* out) {
  async_copy(wsub + SUB_W0_OFF, wbA, 4096 * 2);
  stage_bias(b0, H, biasA);
  wait_async0();
  __syncthreads();

  async_copy(wsub + SUB_WH_OFF, wbB, 16384 * 2);
  stage_bias(bh + 0 * H, H, biasB);
  dense_tanh<1>(ubuf, 32, wbA, biasA, actA);
  wait_async0();
  __syncthreads();

  async_copy(wsub + SUB_WH_OFF + 16384, wbA, 16384 * 2);
  stage_bias(bh + 1 * H, H, biasA);
  dense_tanh<4>(actA, H, wbB, biasB, actB);
  wait_async0();
  __syncthreads();

  async_copy(wsub + SUB_WH_OFF + 2 * 16384, wbB, 16384 * 2);
  stage_bias(bh + 2 * H, H, biasB);
  dense_tanh<4>(actB, H, wbA, biasA, actA);
  wait_async0();
  __syncthreads();

  async_copy(wsub + SUB_WO_OFF, wbA, 2048 * 2);
  stage_bias(bo, out_dim, biasA);
  dense_tanh<4>(actA, H, wbB, biasB, actB);
  wait_async0();
  __syncthreads();

  dense_out(actB, wbA, biasA, out, out_dim);
  __syncthreads();
}

// ---------- prep kernel: f32 weights -> bf16 staged layout in workspace ----------
// subnet id = layer*4 + {0: b/side0, 1: b/side1, 2: a/side0, 3: a/side1}

__global__ __launch_bounds__(THREADS) void prep_kernel(
    const float* __restrict__ Wa0, const float* __restrict__ Wah,
    const float* __restrict__ Wao, const float* __restrict__ Wb0,
    const float* __restrict__ Wbh, const float* __restrict__ Wbo,
    unsigned short* __restrict__ ws) {
  int idx = blockIdx.x * THREADS + threadIdx.x;
  const int total = N_SUBNETS * SUB_TOTAL;
  if (idx >= total) return;
  const int si    = idx / SUB_TOTAL;
  const int r     = idx % SUB_TOTAL;
  const int layer = si >> 2;
  const int q     = si & 3;
  const int side  = q & 1;
  const int isA   = q >> 1;
  const int ls    = layer * 2 + side;
  float v;
  if (r < SUB_WH_OFF) {                       // W0 padded 128x32
    const int row = r >> 5, col = r & 31;
    const int in_dim = isA ? (SPLIT1 + COND) : (SPLIT2 + COND);
    const float* src = (isA ? Wa0 : Wb0) + (size_t)ls * H * in_dim;
    v = (col < in_dim) ? src[row * in_dim + col] : 0.f;
  } else if (r < SUB_WO_OFF) {                // 3 hidden 128x128
    const int rr = r - SUB_WH_OFF;
    const int k  = rr / (H * H);
    const int e  = rr % (H * H);
    const float* src = (isA ? Wah : Wbh) + ((size_t)ls * N_HID + k) * H * H;
    v = src[e];
  } else {                                    // Wo padded 16x128
    const int rr  = r - SUB_WO_OFF;
    const int row = rr >> 7, col = rr & 127;
    const int out_dim = isA ? SPLIT2 : SPLIT1;
    const float* src = (isA ? Wao : Wbo) + (size_t)ls * out_dim * H;
    v = (row < out_dim) ? src[row * H + col] : 0.f;
  }
  ws[idx] = f2bf(v);
}

// ---------- fused full-flow kernel ----------

__global__ __launch_bounds__(THREADS, 1) void flow_kernel(
    const float* __restrict__ xs, const float* __restrict__ ys,
    const unsigned short* __restrict__ ws,
    const float* __restrict__ ba0, const float* __restrict__ bah,
    const float* __restrict__ bao, const float* __restrict__ bb0,
    const float* __restrict__ bbh, const float* __restrict__ bbo,
    const int* __restrict__ perms, float* __restrict__ out_x,
    float* __restrict__ out_jac) {
  __shared__ __align__(16) unsigned short actA[TILE * H];    // 32 KB
  __shared__ __align__(16) unsigned short actB[TILE * H];    // 32 KB
  __shared__ __align__(16) unsigned short wbufA[H * H];      // 32 KB
  __shared__ __align__(16) unsigned short wbufB[H * H];      // 32 KB
  __shared__ __align__(16) unsigned short ubuf[TILE * 32];   // 8 KB
  __shared__ float biasA[H];
  __shared__ float biasB[H];
  __shared__ float xsh[TILE * DIM];
  __shared__ float ysh[TILE * COND];
  __shared__ float stbuf[2 * TILE * 3];   // [0]=s-subnet out, [1]=t-subnet out
  __shared__ float jacS[TILE];

  const int tid  = threadIdx.x;
  const int row0 = blockIdx.x * TILE;

  for (int idx = tid; idx < TILE * DIM; idx += THREADS)
    xsh[idx] = xs[row0 * DIM + idx];
  for (int idx = tid; idx < TILE * COND; idx += THREADS)
    ysh[idx] = ys[row0 * COND + idx];
  if (tid < TILE) jacS[tid] = 0.f;
  __syncthreads();

  const float KC = 2.0f * 0.636f;  // CLAMP * ATAN_SCALE

#pragma unroll 1
  for (int i = 0; i < NUM_LAYERS; ++i) {
    // ---- u2 = [x2, ys], padded to 32 cols ----
    for (int idx = tid; idx < TILE * 32; idx += THREADS) {
      int r = idx >> 5, c = idx & 31;
      float v = 0.f;
      if (c < SPLIT2)              v = xsh[r * DIM + SPLIT1 + c];
      else if (c < SPLIT2 + COND)  v = ysh[r * COND + (c - SPLIT2)];
      ubuf[idx] = f2bf(v);
    }
    __syncthreads();
#pragma unroll 1
    for (int s = 0; s < 2; ++s) {   // s-net then t-net (b family)
      const int ls = i * 2 + s;
      subnet(ws + (size_t)(i * 4 + s) * SUB_TOTAL,
             bb0 + ls * H, bbh + (size_t)ls * N_HID * H, bbo + ls * SPLIT1,
             SPLIT1, ubuf, actA, actB, wbufA, wbufB, biasA, biasB,
             stbuf + s * (TILE * 3));
    }
    // ---- y1 = x1*exp(s2)+t2 ; jac += sum(s2) ----
    if (tid < TILE) {
      float j = jacS[tid];
#pragma unroll
      for (int c = 0; c < SPLIT1; ++c) {
        float sv = KC * atanf(stbuf[tid * 3 + c]);
        xsh[tid * DIM + c] =
            xsh[tid * DIM + c] * __expf(sv) + stbuf[TILE * 3 + tid * 3 + c];
        j += sv;
      }
      jacS[tid] = j;
    }
    __syncthreads();
    // ---- u1 = [y1, ys], padded to 32 cols ----
    for (int idx = tid; idx < TILE * 32; idx += THREADS) {
      int r = idx >> 5, c = idx & 31;
      float v = 0.f;
      if (c < SPLIT1)              v = xsh[r * DIM + c];
      else if (c < SPLIT1 + COND)  v = ysh[r * COND + (c - SPLIT1)];
      ubuf[idx] = f2bf(v);
    }
    __syncthreads();
#pragma unroll 1
    for (int s = 0; s < 2; ++s) {   // s-net then t-net (a family)
      const int ls = i * 2 + s;
      subnet(ws + (size_t)(i * 4 + 2 + s) * SUB_TOTAL,
             ba0 + ls * H, bah + (size_t)ls * N_HID * H, bao + ls * SPLIT2,
             SPLIT2, ubuf, actA, actB, wbufA, wbufB, biasA, biasB,
             stbuf + s * (TILE * 3));
    }
    // ---- y2 = x2*exp(s1)+t1 ; jac += sum(s1) ; permute ----
    if (tid < TILE) {
      float j = jacS[tid];
      float xv[DIM];
#pragma unroll
      for (int d = 0; d < DIM; ++d) xv[d] = xsh[tid * DIM + d];
#pragma unroll
      for (int c = 0; c < SPLIT2; ++c) {
        float sv = KC * atanf(stbuf[tid * 3 + c]);
        xv[SPLIT1 + c] =
            xv[SPLIT1 + c] * __expf(sv) + stbuf[TILE * 3 + tid * 3 + c];
        j += sv;
      }
      jacS[tid] = j;
      float tv[DIM];
#pragma unroll
      for (int d = 0; d < DIM; ++d) tv[d] = xv[perms[i * DIM + d]];
#pragma unroll
      for (int d = 0; d < DIM; ++d) xsh[tid * DIM + d] = tv[d];
    }
    __syncthreads();
  }

  for (int idx = tid; idx < TILE * DIM; idx += THREADS)
    out_x[row0 * DIM + idx] = xsh[idx];
  if (tid < TILE) out_jac[row0 + tid] = jacS[tid];
}

// ---------- entry point ----------

extern "C" void kernel_launch(void* const* d_in, const int* in_sizes, int n_in,
                              void* d_out, int out_size, void* d_ws, size_t ws_size,
                              hipStream_t stream) {
  const float* xs  = (const float*)d_in[0];
  const float* ys  = (const float*)d_in[1];
  const float* Wa0 = (const float*)d_in[2];
  const float* ba0 = (const float*)d_in[3];
  const float* Wah = (const float*)d_in[4];
  const float* bah = (const float*)d_in[5];
  const float* Wao = (const float*)d_in[6];
  const float* bao = (const float*)d_in[7];
  const float* Wb0 = (const float*)d_in[8];
  const float* bb0 = (const float*)d_in[9];
  const float* Wbh = (const float*)d_in[10];
  const float* bbh = (const float*)d_in[11];
  const float* Wbo = (const float*)d_in[12];
  const float* bbo = (const float*)d_in[13];
  const int*  perms = (const int*)d_in[14];

  unsigned short* ws = (unsigned short*)d_ws;
  float* out_x   = (float*)d_out;
  float* out_jac = out_x + (size_t)BATCH * DIM;

  const int prep_total  = N_SUBNETS * SUB_TOTAL;
  const int prep_blocks = (prep_total + THREADS - 1) / THREADS;
  prep_kernel<<<prep_blocks, THREADS, 0, stream>>>(Wa0, Wah, Wao, Wb0, Wbh, Wbo, ws);

  flow_kernel<<<BATCH / TILE, THREADS, 0, stream>>>(
      xs, ys, ws, ba0, bah, bao, bb0, bbh, bbo, perms, out_x, out_jac);
}
